// NearestNeighborLSTM_34772055228498
// MI455X (gfx1250) — matrix operand order repaired and checked by reading, not compiled, not run on hardware
//
#include <hip/hip_runtime.h>
#include <hip/hip_bf16.h>

typedef __attribute__((ext_vector_type(16))) __bf16 v16bf;
typedef __attribute__((ext_vector_type(8)))  float  v8f;

#define BATCH 64
#define TRK   512
#define HID   256
#define ODIM  32
#define KTOT  288      // 32 (x) + 256 (h0)
#define KT    9        // K tiles of 32
#define NT    64       // N tiles of 16 (4*HID / 16)
#define ROWS  (BATCH*TRK)   // 32768

// ---- workspace layout (bytes) ----
#define X_OFF      0                          // 32768*32 f32 = 4 MiB
#define WPACK_OFF  (4u<<20)                   // 9*64*512 bf16 = 589824 B
#define WOUT_OFF   (WPACK_OFF + 9*64*512*2)   // 8*2*512 bf16 = 16384 B
#define BCOMB_OFF  (WOUT_OFF + 8*2*512*2)     // 1024 f32

__device__ __forceinline__ float sigf(float x) {
  return 1.0f / (1.0f + __expf(-x));
}

// -------------------------------------------------------------------------
// Kernel 0: repack weights into WMMA-native bf16 B-tiles.
// B-tile (32x16, bf16): lane L<16 holds col N=L, K=0..7,16..23;
// lane L>=16 holds col N=L-16, K=8..15,24..31 (16 bf16 = 32B/lane).
// -------------------------------------------------------------------------
__global__ void pack_weights(const float* __restrict__ W_ih,
                             const float* __restrict__ b_ih,
                             const float* __restrict__ W_hh,
                             const float* __restrict__ b_hh,
                             const float* __restrict__ W_out,
                             __bf16* __restrict__ wpack,
                             __bf16* __restrict__ woutpack,
                             float* __restrict__ bcomb) {
  const int NW = KT * NT * 512;   // 294912
  const int NO = 8 * 2 * 512;     // 8192
  int idx = blockIdx.x * blockDim.x + threadIdx.x;
  if (idx < NW) {
    int tile = idx >> 9, r = idx & 511;
    int lane = r >> 4, e = r & 15;
    int kt = tile / NT, nt = tile % NT;
    int n = nt * 16 + (lane & 15);
    int k = kt * 32 + ((lane & 16) ? 8 : 0) + (e & 7) + ((e & 8) ? 16 : 0);
    float v = (k < 32) ? W_ih[n * 32 + k] : W_hh[n * HID + (k - 32)];
    wpack[idx] = (__bf16)v;
  } else if (idx < NW + NO) {
    int q = idx - NW;
    int tile = q >> 9, r = q & 511;
    int lane = r >> 4, e = r & 15;
    int kt = tile >> 1, ont = tile & 1;
    int n = ont * 16 + (lane & 15);
    int k = kt * 32 + ((lane & 16) ? 8 : 0) + (e & 7) + ((e & 8) ? 16 : 0);
    woutpack[q] = (__bf16)W_out[n * HID + k];
  } else if (idx < NW + NO + 4 * HID) {
    int q = idx - NW - NO;
    bcomb[q] = b_ih[q] + b_hh[q];
  }
}

// -------------------------------------------------------------------------
// Kernel 1: pairwise top-4 nearest neighbours + 8x4 ReLU embedding.
// One block = 256 tracks of one batch; pos/vel for the batch staged in LDS.
// -------------------------------------------------------------------------
__global__ __launch_bounds__(256)
void neigh_emb(const float* __restrict__ obs1,
               const float* __restrict__ obs2,
               const float* __restrict__ W_emb,
               const float* __restrict__ b_emb,
               float* __restrict__ x) {
  __shared__ float pos[TRK * 2];
  __shared__ float vel[TRK * 2];
  __shared__ float wemb[32];
  __shared__ float bemb[8];

  const int tid = threadIdx.x;
  const int b = blockIdx.x >> 1;
  const int tbase = (blockIdx.x & 1) * 256;

  for (int i = tid; i < TRK * 2; i += 256) {
    float o2 = obs2[b * TRK * 2 + i];
    pos[i] = o2;
    vel[i] = o2 - obs1[b * TRK * 2 + i];
  }
  if (tid < 32) wemb[tid] = W_emb[tid];
  if (tid < 8)  bemb[tid] = b_emb[tid];
  __syncthreads();

  const int t = tbase + tid;
  const float px = pos[2 * t], py = pos[2 * t + 1];
  const float vx = vel[2 * t], vy = vel[2 * t + 1];

  float d0 = 3.0e38f, d1 = 3.0e38f, d2s = 3.0e38f, d3 = 3.0e38f;
  int   i0 = 0, i1 = 0, i2 = 0, i3 = 0;

  for (int j = 0; j < TRK; ++j) {
    if (j == t) continue;
    float dx = pos[2 * j] - px;
    float dy = pos[2 * j + 1] - py;
    float d = dx * dx + dy * dy;           // monotone in norm; same order
    if (d < d3) {                          // strict < keeps lower index on tie
      if (d < d2s) {
        d3 = d2s; i3 = i2;
        if (d < d1) {
          d2s = d1; i2 = i1;
          if (d < d0) { d1 = d0; i1 = i0; d0 = d; i0 = j; }
          else        { d1 = d;  i1 = j; }
        } else { d2s = d; i2 = j; }
      } else { d3 = d; i3 = j; }
    }
  }

  const int row = blockIdx.x * 256 + tid;  // == b*TRK + t
  int nb[4] = {i0, i1, i2, i3};
#pragma unroll
  for (int n = 0; n < 4; ++n) {
    int j = nb[n];
    float f0 = pos[2 * j] - px;
    float f1 = pos[2 * j + 1] - py;
    float f2 = vel[2 * j] - vx;
    float f3 = vel[2 * j + 1] - vy;
#pragma unroll
    for (int o = 0; o < 8; ++o) {
      float e = wemb[o * 4 + 0] * f0 + wemb[o * 4 + 1] * f1 +
                wemb[o * 4 + 2] * f2 + wemb[o * 4 + 3] * f3 + bemb[o];
      x[row * ODIM + n * 8 + o] = fmaxf(e, 0.0f);
    }
  }
}

// -------------------------------------------------------------------------
// Kernel 2: fused LSTM-cell step + output projection on WMMA.
// 2 waves/block; each wave owns a 16-row M-panel.
//   phase 1: stage A = [x | h0] (16x288) as bf16 A-layout tiles in LDS
//   phase 2: per hidden 16-col tile: 4 gate accs (i,f,g,o) x 9 WMMAs,
//            fuse biases + sigmoid/tanh + c0 in registers, h -> LDS (A-layout)
//   phase 3: out(16x32) = h(16x256) @ WoutPack : 2 x 8 WMMAs + b_out
// -------------------------------------------------------------------------
__global__ __launch_bounds__(64)
void lstm_gemm(const float* __restrict__ x,
               const float* __restrict__ h0,
               const float* __restrict__ c0,
               const __bf16* __restrict__ wpack,
               const __bf16* __restrict__ woutpack,
               const float* __restrict__ bcomb,
               const float* __restrict__ b_out,
               float* __restrict__ out) {
  __shared__ __align__(32) __bf16 Asm[2][KT * 512];  // 18 KiB
  __shared__ __align__(32) __bf16 Hsm[2][8 * 512];   // 16 KiB

  const int tid = threadIdx.x;
  const int w = tid >> 5;
  const int lane = tid & 31;
  const int rowBase = blockIdx.x * 32 + w * 16;

  // ---- phase 1: A-panel to LDS (ISA 16-bit A-layout) ----
  for (int i = lane; i < 16 * KTOT; i += 32) {
    int m = i / KTOT, k = i % KTOT;
    float v = (k < 32) ? x[(rowBase + m) * ODIM + k]
                       : h0[(size_t)(rowBase + m) * HID + (k - 32)];
    int kt = k >> 5, kl = k & 31;
    int lp = m + ((kl & 8) ? 16 : 0);            // lane within tile
    int e  = (kl & 7) + ((kl & 16) ? 8 : 0);     // element within lane
    Asm[w][kt * 512 + lp * 16 + e] = (__bf16)v;
  }
  __syncthreads();

  const v16bf* Av = (const v16bf*)&Asm[w][0];
  const v16bf* Bv = (const v16bf*)wpack;
  const int mofs = (lane < 16) ? 0 : 8;
  const int colin = lane & 15;

  // ---- phase 2: gates + LSTM activations ----
  for (int ht = 0; ht < 16; ++ht) {
    v8f ai = {}, af = {}, ag = {}, ao = {};
#pragma unroll
    for (int kt = 0; kt < KT; ++kt) {
      v16bf a = Av[kt * 32 + lane];
      const v16bf* bb = &Bv[(size_t)(kt * NT + ht) * 32 + lane];
      if (kt + 1 < KT)
        __builtin_prefetch(&Bv[(size_t)((kt + 1) * NT + ht) * 32 + lane], 0, 1);
      v16bf b0 = bb[0];            // gate i : nt = ht
      v16bf b1 = bb[16 * 32];      // gate f : nt = 16 + ht
      v16bf b2 = bb[32 * 32];      // gate g : nt = 32 + ht
      v16bf b3 = bb[48 * 32];      // gate o : nt = 48 + ht
      ai = __builtin_amdgcn_wmma_f32_16x16x32_bf16(false, a, false, b0, (short)0, ai, false, false);
      af = __builtin_amdgcn_wmma_f32_16x16x32_bf16(false, a, false, b1, (short)0, af, false, false);
      ag = __builtin_amdgcn_wmma_f32_16x16x32_bf16(false, a, false, b2, (short)0, ag, false, false);
      ao = __builtin_amdgcn_wmma_f32_16x16x32_bf16(false, a, false, b3, (short)0, ao, false, false);
    }
    const int col = ht * 16 + colin;
    const float bi = bcomb[col];
    const float bf = bcomb[HID + col];
    const float bg = bcomb[2 * HID + col];
    const float bo = bcomb[3 * HID + col];
    const int kt2 = col >> 5, kl = col & 31;
    const int e = (kl & 7) + ((kl & 16) ? 8 : 0);
    const int lpadd = (kl & 8) ? 16 : 0;
#pragma unroll
    for (int r = 0; r < 8; ++r) {
      int m = rowBase + r + mofs;
      float iv = ai[r] + bi;
      float fv = af[r] + bf;
      float gv = ag[r] + bg;
      float ov = ao[r] + bo;
      float c0v = c0[(size_t)m * HID + col];
      float cc = sigf(fv) * c0v + sigf(iv) * tanhf(gv);
      float hh = sigf(ov) * tanhf(cc);
      int lp = (r + mofs) + lpadd;
      Hsm[w][kt2 * 512 + lp * 16 + e] = (__bf16)hh;
    }
  }
  __syncthreads();

  // ---- phase 3: output projection ----
  const v16bf* Hv = (const v16bf*)&Hsm[w][0];
  const v16bf* Wov = (const v16bf*)woutpack;
#pragma unroll
  for (int ont = 0; ont < 2; ++ont) {
    v8f acc = {};
#pragma unroll
    for (int kt = 0; kt < 8; ++kt) {
      v16bf a = Hv[kt * 32 + lane];
      v16bf b = Wov[(kt * 2 + ont) * 32 + lane];
      acc = __builtin_amdgcn_wmma_f32_16x16x32_bf16(false, a, false, b, (short)0, acc, false, false);
    }
    const int col = ont * 16 + colin;
    const float bov = b_out[col];
#pragma unroll
    for (int r = 0; r < 8; ++r) {
      int m = rowBase + r + mofs;
      out[m * ODIM + col] = acc[r] + bov;
    }
  }
}

// -------------------------------------------------------------------------
extern "C" void kernel_launch(void* const* d_in, const int* in_sizes, int n_in,
                              void* d_out, int out_size, void* d_ws, size_t ws_size,
                              hipStream_t stream) {
  const float* obs1  = (const float*)d_in[0];
  const float* obs2  = (const float*)d_in[1];
  // d_in[2] = h0, d_in[3] = c0
  const float* h0    = (const float*)d_in[2];
  const float* c0    = (const float*)d_in[3];
  const float* W_emb = (const float*)d_in[4];
  const float* b_emb = (const float*)d_in[5];
  const float* W_ih  = (const float*)d_in[6];
  const float* b_ih  = (const float*)d_in[7];
  const float* W_hh  = (const float*)d_in[8];
  const float* b_hh  = (const float*)d_in[9];
  const float* W_out = (const float*)d_in[10];
  const float* b_out = (const float*)d_in[11];
  float* out = (float*)d_out;

  char* ws = (char*)d_ws;
  float*  xbuf     = (float*)(ws + X_OFF);
  __bf16* wpack    = (__bf16*)(ws + WPACK_OFF);
  __bf16* woutpack = (__bf16*)(ws + WOUT_OFF);
  float*  bcomb    = (float*)(ws + BCOMB_OFF);

  // 0) weight repack (tiny; runs every launch — deterministic)
  {
    int total = KT * NT * 512 + 8 * 2 * 512 + 4 * HID;
    int blocks = (total + 255) / 256;
    pack_weights<<<blocks, 256, 0, stream>>>(W_ih, b_ih, W_hh, b_hh, W_out,
                                             wpack, woutpack, bcomb);
  }
  // 1) neighbour search + embedding -> x
  neigh_emb<<<BATCH * 2, 256, 0, stream>>>(obs1, obs2, W_emb, b_emb, xbuf);

  // 2) fused LSTM + output projection (WMMA)
  lstm_gemm<<<ROWS / 32, 64, 0, stream>>>(xbuf, h0, c0, wpack, woutpack,
                                          bcomb, b_out, out);
}
